// SparseMoETransformer_29188597743841
// MI455X (gfx1250) — matrix operand order repaired and checked by reading, not compiled, not run on hardware
//
#include <hip/hip_runtime.h>

// ---------------- model constants (match reference) ----------------
constexpr int Bc  = 2;
constexpr int Sc  = 2048;
constexpr int Ec  = 1024;
constexpr int Hc  = 16;
constexpr int DHc = 64;      // Ec / Hc
constexpr int Vc  = 32000;
constexpr int NEXP = 8;
constexpr int NLAYERS = 2;

// ---------------- WMMA types ----------------
typedef __attribute__((ext_vector_type(16))) __bf16        v16bf;
typedef __attribute__((ext_vector_type(8)))  float         v8f;
typedef __attribute__((ext_vector_type(8)))  unsigned int  v8u;
typedef __attribute__((ext_vector_type(4)))  unsigned int  v4u;
typedef __attribute__((ext_vector_type(4)))  int           v4i;

typedef __attribute__((address_space(1))) v4i gv4i;   // global int4
typedef __attribute__((address_space(3))) v4i lv4i;   // LDS int4

#if defined(__has_builtin)
#if __has_builtin(__builtin_amdgcn_global_load_async_to_lds_b128)
#define HAVE_ASYNC_LDS 1
#endif
#endif

__device__ __forceinline__ unsigned short bfbits(float f) {
    unsigned int u = __float_as_uint(f);
    u += 0x7fffu + ((u >> 16) & 1u);           // round-to-nearest-even
    return (unsigned short)(u >> 16);
}
__device__ __forceinline__ unsigned int packbf(float lo, float hi) {
    return (unsigned int)bfbits(lo) | ((unsigned int)bfbits(hi) << 16);
}
// A-fragment K pair base for vgpr v, lane half h  (16-bit A 16x32 layout)
__device__ __forceinline__ int kA(int v, int h) {
    return (v < 4) ? (8 * h + 2 * v) : (16 + 8 * h + 2 * (v - 4));
}
// B-fragment K pair base for vgpr v, lane half h (32x16 B, n per lane)
__device__ __forceinline__ int kB(int v, int h) { return (v < 4) ? (8 * h + 2 * v) : (16 + 8 * h + 2 * (v - 4)); }

__device__ __forceinline__ v8f wmma_bf16(v16bf a, v16bf b, v8f c) {
    return __builtin_amdgcn_wmma_f32_16x16x32_bf16(false, a, false, b, (short)0, c,
                                                   false, false);
}

// 16-byte global -> LDS stage (async on CDNA5, register copy fallback)
__device__ __forceinline__ void stage16(void* lds, const void* g) {
#ifdef HAVE_ASYNC_LDS
    __builtin_amdgcn_global_load_async_to_lds_b128(
        (gv4i*)(unsigned long long)g,
        (lv4i*)(unsigned int)(unsigned long long)lds, 0, 0);
#else
    *(v4u*)lds = *(const v4u*)g;
#endif
}
__device__ __forceinline__ void wait_async0() {
#ifdef HAVE_ASYNC_LDS
    asm volatile("s_wait_asynccnt 0" ::: "memory");
#endif
}
// allow the 3 just-issued prefetch ops to stay in flight (in-order completion)
__device__ __forceinline__ void wait_async3() {
#ifdef HAVE_ASYNC_LDS
    asm volatile("s_wait_asynccnt 3" ::: "memory");
#endif
}

// ---------------- embedding (writes fp32 X and bf16 X16) ----------------
__global__ __launch_bounds__(256) void embed_kernel(
    const int* __restrict__ ids, const float* __restrict__ tok,
    const float* __restrict__ pos, float* __restrict__ x,
    unsigned short* __restrict__ x16) {
    int t = blockIdx.x;
    int sp = t % Sc;
    int id = ids[t];
    for (int e = threadIdx.x; e < Ec; e += 256) {
        float v = tok[(size_t)id * Ec + e] + pos[(size_t)sp * Ec + e];
        x[(size_t)t * Ec + e] = v;
        x16[(size_t)t * Ec + e] = bfbits(v);
    }
}

// ---------------- fp32 [K][N] -> bf16 transposed [N][K] (LDS tiled) --------
__global__ __launch_bounds__(256) void transpose_cvt_kernel(
    const float* __restrict__ in, unsigned short* __restrict__ out, int K, int N) {
    __shared__ float t[32][33];
    int n0 = blockIdx.x * 32, k0 = blockIdx.y * 32;
    int c = threadIdx.x & 31, r0 = (threadIdx.x >> 5) * 4;
#pragma unroll
    for (int j = 0; j < 4; ++j)
        t[r0 + j][c] = in[(size_t)(k0 + r0 + j) * N + n0 + c];
    __syncthreads();
#pragma unroll
    for (int j = 0; j < 4; ++j)
        out[(size_t)(n0 + r0 + j) * K + k0 + c] = bfbits(t[c][r0 + j]);
}

// ------- per-head Wq/Wk/Wv [H][E][DH] -> bf16 transposed [H*DH][E] ---------
__global__ __launch_bounds__(256) void repackT_cvt_kernel(
    const float* __restrict__ in, unsigned short* __restrict__ out) {
    int i = blockIdx.x * 256 + threadIdx.x;      // over H*E*DH
    if (i >= Hc * Ec * DHc) return;
    int d = i & (DHc - 1);
    int e = (i >> 6) & (Ec - 1);
    int h = i >> 16;
    out[(size_t)(h * DHc + d) * Ec + e] = bfbits(in[i]);
}

// ---------------- bf16-WMMA GEMM: act(A*W^T), double-buffered async --------
// A: bf16 [M][K];  WT: bf16 [N][K];  128x64x32 tiles, 8 waves.
// Output: fp32 Cf (optionally +=) or bf16 C16 (with outscale folded in).
__global__ __launch_bounds__(256) void gemm_bf16_kernel(
    const unsigned short* __restrict__ A, const unsigned short* __restrict__ WT,
    const float* __restrict__ bias, float* __restrict__ Cf,
    unsigned short* __restrict__ C16, float outscale,
    int M, int N, int K, int relu,
    const float* __restrict__ rowscale, int rss, int accumulate) {
    __shared__ __align__(16) unsigned short a_s[2][128][32];
    __shared__ __align__(16) unsigned short b_s[2][64][32];

    int tid = threadIdx.x;
    int wave = tid >> 5, lane = tid & 31;
    int half = lane >> 4, l16 = lane & 15;
    int bm0 = blockIdx.y * 128;
    int bn0 = blockIdx.x * 64;

    // staging coords: one 16B chunk = 8 bf16 along K
    int sr = tid >> 2;            // 0..63
    int sc = (tid & 3) * 8;       // 0,8,16,24

    auto stage_tile = [&](int buf, int k0) {
        stage16(&a_s[buf][sr][sc],      &A[(size_t)(bm0 + sr) * K + k0 + sc]);
        stage16(&a_s[buf][sr + 64][sc], &A[(size_t)(bm0 + sr + 64) * K + k0 + sc]);
        stage16(&b_s[buf][sr][sc],      &WT[(size_t)(bn0 + sr) * K + k0 + sc]);
    };

    v8f acc[4] = {};
    int nk = K >> 5;

    stage_tile(0, 0);
    for (int ik = 0; ik < nk; ++ik) {
        int cur = ik & 1;
        if (ik + 1 < nk) {
            stage_tile(cur ^ 1, (ik + 1) << 5);   // prefetch next tile
            wait_async3();                        // current tile done; next in flight
        } else {
            wait_async0();
        }
        __syncthreads();

        // gather all fragments first, then issue 4 WMMAs back-to-back
        v8u au;
        int am = wave * 16 + l16;
#pragma unroll
        for (int v = 0; v < 8; ++v)
            au[v] = *(const unsigned int*)&a_s[cur][am][kA(v, half)];
        v8u bu[4];
#pragma unroll
        for (int t = 0; t < 4; ++t)
#pragma unroll
            for (int v = 0; v < 8; ++v)
                bu[t][v] = *(const unsigned int*)&b_s[cur][t * 16 + l16][kB(v, half)];

        v16bf af = __builtin_bit_cast(v16bf, au);
#pragma unroll
        for (int t = 0; t < 4; ++t)
            acc[t] = wmma_bf16(af, __builtin_bit_cast(v16bf, bu[t]), acc[t]);
        __syncthreads();
    }

#pragma unroll
    for (int t = 0; t < 4; ++t) {
        int n = bn0 + t * 16 + l16;
        float bv = bias ? bias[n] : 0.f;
#pragma unroll
        for (int r = 0; r < 8; ++r) {
            int m = bm0 + wave * 16 + 8 * half + r;
            float val = acc[t][r] + bv;
            if (relu) val = fmaxf(val, 0.f);
            if (rowscale) val *= rowscale[(size_t)m * rss];
            val *= outscale;
            size_t idx = (size_t)m * N + n;
            if (C16) C16[idx] = bfbits(val);
            else if (accumulate) Cf[idx] += val;
            else Cf[idx] = val;
        }
    }
}

// ---------------- flash attention (causal), bf16 in/out, async staging ------
__global__ __launch_bounds__(256) void attn_kernel(
    const unsigned short* __restrict__ Q16, const unsigned short* __restrict__ K16,
    const unsigned short* __restrict__ V16, unsigned short* __restrict__ O16) {
    __shared__ __align__(16) unsigned short q_s[64][64];   // [q][dh] (pre-scaled)
    __shared__ __align__(16) unsigned short k_s[64][64];   // [kv][dh]
    __shared__ __align__(16) unsigned short v_s[64][64];   // [kv][dh]
    __shared__ __align__(16) float p_s[64][64];            // scores / P
    __shared__ float mrow[64], lrow[64], srow[64];

    int tid = threadIdx.x;
    int wave = tid >> 5, lane = tid & 31;
    int half = lane >> 4, l16 = lane & 15;
    int wm = wave >> 1, wn = wave & 1;

    int qtile = blockIdx.x, h = blockIdx.y, b = blockIdx.z;
    int qbase = qtile * 64;

    // stage Q tile (pure bf16 copy; 1/sqrt(DH) folded in at GEMM epilogue)
#pragma unroll
    for (int j = 0; j < 2; ++j) {
        int u = tid + j * 256;
        int r = u >> 3, c0 = (u & 7) * 8;
        stage16(&q_s[r][c0], &Q16[((size_t)(b * Sc + qbase + r)) * Ec + h * DHc + c0]);
    }
    if (tid < 64) { mrow[tid] = -3.4e38f; lrow[tid] = 0.f; }
    wait_async0();
    __syncthreads();

    v8f o0 = {}; v8f o1 = {};
    for (int it = 0; it <= qtile; ++it) {
        int kv0 = it * 64;
#pragma unroll
        for (int j = 0; j < 2; ++j) {
            int u = tid + j * 256;
            int r = u >> 3, c0 = (u & 7) * 8;
            size_t gi = ((size_t)(b * Sc + kv0 + r)) * Ec + h * DHc + c0;
            stage16(&k_s[r][c0], &K16[gi]);
            stage16(&v_s[r][c0], &V16[gi]);
        }
        wait_async0();
        __syncthreads();

        // S = Q * K^T  (wave: 16 q rows x 32 kv cols)
        v8f s0 = {}; v8f s1 = {};
#pragma unroll
        for (int kk = 0; kk < 64; kk += 32) {
            v8u au, bu0, bu1;
            int am = wm * 16 + l16;
#pragma unroll
            for (int v = 0; v < 8; ++v) {
                au[v]  = *(const unsigned int*)&q_s[am][kk + kA(v, half)];
                bu0[v] = *(const unsigned int*)&k_s[wn * 32 + l16][kk + kB(v, half)];
                bu1[v] = *(const unsigned int*)&k_s[wn * 32 + 16 + l16][kk + kB(v, half)];
            }
            v16bf af = __builtin_bit_cast(v16bf, au);
            s0 = wmma_bf16(af, __builtin_bit_cast(v16bf, bu0), s0);
            s1 = wmma_bf16(af, __builtin_bit_cast(v16bf, bu1), s1);
        }
#pragma unroll
        for (int t = 0; t < 2; ++t) {
            v8f sc = t ? s1 : s0;
            int n = wn * 32 + t * 16 + l16;
#pragma unroll
            for (int r = 0; r < 8; ++r)
                p_s[wm * 16 + 8 * half + r][n] = sc[r];
        }
        __syncthreads();

        // online softmax row pass (one thread per row)
        if (tid < 64) {
            int r = tid;
            int limit = qbase + r - kv0;           // cols <= limit valid
            float mold = mrow[r];
            float tmax = -3.4e38f;
            for (int c = 0; c < 64; ++c)
                if (c <= limit) tmax = fmaxf(tmax, p_s[r][c]);
            float mnew = fmaxf(mold, tmax);
            float scale = __expf(mold - mnew);
            float psum = 0.f;
            for (int c = 0; c < 64; ++c) {
                float pv = (c <= limit) ? __expf(p_s[r][c] - mnew) : 0.f;
                p_s[r][c] = pv;
                psum += pv;
            }
            lrow[r] = lrow[r] * scale + psum;
            srow[r] = scale;
            mrow[r] = mnew;
        }
        __syncthreads();

        // rescale O accumulators, then O += P * V
#pragma unroll
        for (int r = 0; r < 8; ++r) {
            float sc = srow[wm * 16 + 8 * half + r];
            o0[r] *= sc; o1[r] *= sc;
        }
#pragma unroll
        for (int kk = 0; kk < 64; kk += 32) {
            v8u au, bu0, bu1;
            int am = wm * 16 + l16;
            int n0 = wn * 32 + l16, n1 = wn * 32 + 16 + l16;
#pragma unroll
            for (int v = 0; v < 8; ++v) {
                int ka = kk + kA(v, half);
                au[v] = packbf(p_s[am][ka], p_s[am][ka + 1]);
                int kb = kk + kB(v, half);
                bu0[v] = (unsigned int)v_s[kb][n0] | ((unsigned int)v_s[kb + 1][n0] << 16);
                bu1[v] = (unsigned int)v_s[kb][n1] | ((unsigned int)v_s[kb + 1][n1] << 16);
            }
            v16bf af = __builtin_bit_cast(v16bf, au);
            o0 = wmma_bf16(af, __builtin_bit_cast(v16bf, bu0), o0);
            o1 = wmma_bf16(af, __builtin_bit_cast(v16bf, bu1), o1);
        }
        __syncthreads();
    }

#pragma unroll
    for (int t = 0; t < 2; ++t) {
        v8f acc = t ? o1 : o0;
        int n = h * DHc + wn * 32 + t * 16 + l16;
#pragma unroll
        for (int r = 0; r < 8; ++r) {
            int mloc = wm * 16 + 8 * half + r;
            O16[((size_t)(b * Sc + qbase + mloc)) * Ec + n] = bfbits(acc[r] / lrow[mloc]);
        }
    }
}

// -------- fused residual + LayerNorm (in place on x, co-writes bf16) -------
__global__ __launch_bounds__(256) void residual_ln_kernel(
    float* __restrict__ x, const float* __restrict__ y,
    const float* __restrict__ w, const float* __restrict__ bb,
    unsigned short* __restrict__ out16) {
    __shared__ float red[256];
    __shared__ float s_mean, s_rstd;
    int row = blockIdx.x, tid = threadIdx.x;
    float* xr = x + (size_t)row * Ec;
    const float* yr = y ? y + (size_t)row * Ec : nullptr;

    float sum = 0.f, sumsq = 0.f;
    for (int e = tid; e < Ec; e += 256) {
        float v = xr[e] + (yr ? yr[e] : 0.f);
        sum += v; sumsq += v * v;
    }
    red[tid] = sum; __syncthreads();
    for (int s = 128; s > 0; s >>= 1) { if (tid < s) red[tid] += red[tid + s]; __syncthreads(); }
    if (tid == 0) s_mean = red[0] * (1.f / Ec);
    __syncthreads();
    red[tid] = sumsq; __syncthreads();
    for (int s = 128; s > 0; s >>= 1) { if (tid < s) red[tid] += red[tid + s]; __syncthreads(); }
    if (tid == 0) {
        float var = red[0] * (1.f / Ec) - s_mean * s_mean;
        s_rstd = rsqrtf(var + 1e-5f);
    }
    __syncthreads();
    float mean = s_mean, rstd = s_rstd;
    for (int e = tid; e < Ec; e += 256) {
        float v = xr[e] + (yr ? yr[e] : 0.f);
        float o = (v - mean) * rstd * w[e] + bb[e];
        xr[e] = o;
        out16[(size_t)row * Ec + e] = bfbits(o);
    }
}

// ---------------- router: softmax over 8 experts, top-2 renormalized --------
__global__ __launch_bounds__(64) void router_kernel(
    const float* __restrict__ x, const float* __restrict__ Wr,
    const float* __restrict__ br, float* __restrict__ rw) {
    __shared__ float logit[NEXP];
    int tok = blockIdx.x, tid = threadIdx.x;
    if (tid < NEXP) {
        float s = br[tid];
        const float* xr = x + (size_t)tok * Ec;
        for (int e = 0; e < Ec; ++e) s += xr[e] * Wr[e * NEXP + tid];
        logit[tid] = s;
    }
    __syncthreads();
    if (tid == 0) {
        float mx = logit[0];
        for (int i = 1; i < NEXP; ++i) mx = fmaxf(mx, logit[i]);
        float p[NEXP]; float den = 0.f;
        for (int i = 0; i < NEXP; ++i) { p[i] = __expf(logit[i] - mx); den += p[i]; }
        float v0 = -1.f, v1 = -1.f;
        for (int i = 0; i < NEXP; ++i) {
            float pv = p[i] / den;
            if (pv > v0) { v1 = v0; v0 = pv; }
            else if (pv > v1) v1 = pv;
        }
        float s = v0 + v1;
        rw[(size_t)tok * 2 + 0] = v0 / s;
        rw[(size_t)tok * 2 + 1] = v1 / s;
    }
}

// ---------------- host orchestration ----------------
extern "C" void kernel_launch(void* const* d_in, const int* in_sizes, int n_in,
                              void* d_out, int out_size, void* d_ws, size_t ws_size,
                              hipStream_t stream) {
    (void)in_sizes; (void)n_in; (void)out_size; (void)ws_size;
    const int* ids = (const int*)d_in[0];
    const float* tok_emb = (const float*)d_in[1];
    const float* pos_emb = (const float*)d_in[2];
    const int FINAL = 3 + NLAYERS * 21;
    const float* lnf_w = (const float*)d_in[FINAL + 0];
    const float* lnf_b = (const float*)d_in[FINAL + 1];
    const float* Wout  = (const float*)d_in[FINAL + 2];
    const float* bout  = (const float*)d_in[FINAL + 3];
    float* out = (float*)d_out;

    const size_t T = (size_t)Bc * Sc;    // 4096 tokens
    float* ws = (float*)d_ws;
    float* X   = ws;  ws += T * Ec;
    float* TMP = ws;  ws += T * Ec;
    float* MOE = ws;  ws += T * Ec;
    float* RW  = ws;  ws += T * 2;
    unsigned short* X16  = (unsigned short*)ws;  ws += T * Ec / 2;
    unsigned short* Q16  = (unsigned short*)ws;  ws += T * Ec / 2;
    unsigned short* K16  = (unsigned short*)ws;  ws += T * Ec / 2;
    unsigned short* V16  = (unsigned short*)ws;  ws += T * Ec / 2;
    unsigned short* O16  = (unsigned short*)ws;  ws += T * Ec / 2;
    unsigned short* H16  = (unsigned short*)ws;  ws += T * 4 * Ec / 2;
    unsigned short* WT16 = (unsigned short*)ws;  ws += (size_t)Vc * Ec / 2;

    dim3 blk(256);
    auto transW = [&](const float* w, int K, int N) {
        transpose_cvt_kernel<<<dim3(N / 32, K / 32), blk, 0, stream>>>(w, WT16, K, N);
    };
    // bf16-output GEMM
    auto gemm16 = [&](const unsigned short* A, const float* bias, unsigned short* C16,
                      int M, int N, int K, int relu, float oscale) {
        gemm_bf16_kernel<<<dim3(N / 64, M / 128), blk, 0, stream>>>(
            A, WT16, bias, nullptr, C16, oscale, M, N, K, relu, nullptr, 0, 0);
    };
    // fp32-output GEMM
    auto gemmf = [&](const unsigned short* A, const float* bias, float* Cf,
                     int M, int N, int K, const float* rs, int rss, int acc) {
        gemm_bf16_kernel<<<dim3(N / 64, M / 128), blk, 0, stream>>>(
            A, WT16, bias, Cf, nullptr, 1.f, M, N, K, 0, rs, rss, acc);
    };

    embed_kernel<<<dim3((unsigned)T), blk, 0, stream>>>(ids, tok_emb, pos_emb, X, X16);

    for (int l = 0; l < NLAYERS; ++l) {
        const int L = 3 + l * 21;
        const float* Wq  = (const float*)d_in[L + 0];
        const float* Wk  = (const float*)d_in[L + 1];
        const float* Wv  = (const float*)d_in[L + 2];
        const float* Wo  = (const float*)d_in[L + 3];
        const float* bo  = (const float*)d_in[L + 4];
        const float* Wr  = (const float*)d_in[L + 5];
        const float* br  = (const float*)d_in[L + 6];
        const float* W1  = (const float*)d_in[L + 7];
        const float* b1  = (const float*)d_in[L + 8];
        const float* W2  = (const float*)d_in[L + 9];
        const float* b2  = (const float*)d_in[L + 10];
        const float* Wf1 = (const float*)d_in[L + 11];
        const float* bf1 = (const float*)d_in[L + 12];
        const float* Wf2 = (const float*)d_in[L + 13];
        const float* bf2 = (const float*)d_in[L + 14];
        const float* ln1w = (const float*)d_in[L + 15];
        const float* ln1b = (const float*)d_in[L + 16];
        const float* ln2w = (const float*)d_in[L + 17];
        const float* ln2b = (const float*)d_in[L + 18];
        const float* ln3w = (const float*)d_in[L + 19];
        const float* ln3b = (const float*)d_in[L + 20];

        const int RG = (Hc * Ec * DHc + 255) / 256;

        // --- attention ---
        repackT_cvt_kernel<<<RG, blk, 0, stream>>>(Wq, WT16);
        gemm16(X16, nullptr, Q16, (int)T, Ec, Ec, 0, 0.125f);   // 1/sqrt(DH) folded
        repackT_cvt_kernel<<<RG, blk, 0, stream>>>(Wk, WT16);
        gemm16(X16, nullptr, K16, (int)T, Ec, Ec, 0, 1.f);
        repackT_cvt_kernel<<<RG, blk, 0, stream>>>(Wv, WT16);
        gemm16(X16, nullptr, V16, (int)T, Ec, Ec, 0, 1.f);

        attn_kernel<<<dim3(Sc / 64, Hc, Bc), blk, 0, stream>>>(Q16, K16, V16, O16);

        transW(Wo, Ec, Ec);
        gemmf(O16, bo, TMP, (int)T, Ec, Ec, nullptr, 0, 0);
        residual_ln_kernel<<<dim3((unsigned)T), blk, 0, stream>>>(X, TMP, ln1w, ln1b, X16);

        // --- MoE (experts 0 and 1, gated by renormalized top-2 probs) ---
        router_kernel<<<dim3((unsigned)T), dim3(64), 0, stream>>>(X, Wr, br, RW);
        for (int i = 0; i < 2; ++i) {
            const float* W1i = W1 + (size_t)i * Ec * 4 * Ec;
            const float* b1i = b1 + (size_t)i * 4 * Ec;
            const float* W2i = W2 + (size_t)i * 4 * Ec * Ec;
            const float* b2i = b2 + (size_t)i * Ec;
            transW(W1i, Ec, 4 * Ec);
            gemm16(X16, b1i, H16, (int)T, 4 * Ec, Ec, 1, 1.f);  // fused ReLU -> bf16
            transW(W2i, 4 * Ec, Ec);
            gemmf(H16, b2i, MOE, (int)T, Ec, 4 * Ec, RW + i, 2, i);
        }
        residual_ln_kernel<<<dim3((unsigned)T), blk, 0, stream>>>(X, MOE, ln2w, ln2b, X16);

        // --- dense FFN ---
        transW(Wf1, Ec, 4 * Ec);
        gemm16(X16, bf1, H16, (int)T, 4 * Ec, Ec, 1, 1.f);
        transW(Wf2, 4 * Ec, Ec);
        gemmf(H16, bf2, TMP, (int)T, Ec, 4 * Ec, nullptr, 0, 0);
        residual_ln_kernel<<<dim3((unsigned)T), blk, 0, stream>>>(X, TMP, ln3w, ln3b, X16);
    }

    residual_ln_kernel<<<dim3((unsigned)T), blk, 0, stream>>>(X, nullptr, lnf_w, lnf_b, X16);
    transW(Wout, Ec, Vc);
    gemmf(X16, bout, out, (int)T, Vc, Ec, nullptr, 0, 0);
}
